// Model_23579370455462
// MI455X (gfx1250) — matrix operand order repaired and checked by reading, not compiled, not run on hardware
//
#include <hip/hip_runtime.h>
#include <hip/hip_bf16.h>
#include <stdint.h>

// Problem constants (from reference)
#define NU_N   100000
#define NR_N   50000
#define E_N    1000000
#define EP_N   500000
#define EN_N   500000

typedef __attribute__((ext_vector_type(16))) __bf16          v16bf;
typedef __attribute__((ext_vector_type(8)))  float           v8f;
typedef __attribute__((ext_vector_type(16))) unsigned short  v16us;

__device__ __forceinline__ unsigned short f32_to_bf16_rne(float f) {
    unsigned int u = __float_as_uint(f);
    unsigned int r = u + 0x7FFFu + ((u >> 16) & 1u);
    return (unsigned short)(r >> 16);
}
__device__ __forceinline__ float bf16_bits_to_f32(unsigned short h) {
    return __uint_as_float(((unsigned int)h) << 16);
}

// ---------------------------------------------------------------------------
// WMMA GEMM: C[M,N] = (rowscale ? diag(rowscale) : I) * A[M,K] @ W[K,N] + bias
// Split-bf16 (hi/lo) -> 3x v_wmma_f32_16x16x32_bf16 per 16x16 tile, f32 acc.
// Block = 256 thr = 8 waves; wave -> 16 rows x 64 cols; grid.y tiles N by 64.
// W staged in LDS pre-swizzled into per-lane fragment layout (hi+lo planes).
// WMMAs issued pass-by-pass over 4 independent accumulators to avoid
// back-to-back same-destination chains (bf16 WMMA pipeline depth ~5).
// ---------------------------------------------------------------------------
template <int K, int N>
__global__ __launch_bounds__(256) void gemm_bias_wmma(
    const float* __restrict__ A, const float* __restrict__ rowscale,
    const float* __restrict__ W, const float* __restrict__ bias,
    float* __restrict__ C, int M)
{
    constexpr int KT = K / 32;                 // k-steps of 32
    __shared__ unsigned short ldsW[2 * K * 64]; // [hi plane | lo plane]

    const int tid  = threadIdx.x;
    const int lane = tid & 31;
    const int wave = tid >> 5;
    const int nb0  = blockIdx.y * 64;

    // Stage W (this block's 64 columns) into fragment-swizzled LDS layout.
    // idx -> (t = ktile, u = ntile, l = lane, j = frag element)
    for (int idx = tid; idx < K * 64; idx += 256) {
        int j = idx & 15;
        int l = (idx >> 4) & 31;
        int u = (idx >> 9) & 3;
        int t = idx >> 11;
        int n = nb0 + u * 16 + (l & 15);
        int k = t * 32 + ((l & 16) ? 8 : 0) + (j < 8 ? j : j + 8);   // B frag: K = kb + (j<8 ? j : j+8)
        float w = W[k * N + n];
        unsigned short hb = f32_to_bf16_rne(w);
        float lo = w - bf16_bits_to_f32(hb);
        ldsW[idx]           = hb;
        ldsW[K * 64 + idx]  = f32_to_bf16_rne(lo);
    }
    __syncthreads();

    const int mrow0 = blockIdx.x * 128 + wave * 16;
    const int arow  = min(mrow0 + (lane & 15), M - 1);  // clamp for edge tile
    const float s   = rowscale ? rowscale[arow] : 1.0f;
    const int koff  = (lane & 16) ? 8 : 0;              // A frag K offset

    v8f acc[4] = {};

    for (int t = 0; t < KT; ++t) {
        const int kk = t * 32;
        const float* ap = A + (size_t)arow * K + kk + koff;
        float4 f0 = *(const float4*)(ap + 0);
        float4 f1 = *(const float4*)(ap + 4);
        float4 f2 = *(const float4*)(ap + 16);
        float4 f3 = *(const float4*)(ap + 20);
        float av[16] = { f0.x, f0.y, f0.z, f0.w, f1.x, f1.y, f1.z, f1.w,
                         f2.x, f2.y, f2.z, f2.w, f3.x, f3.y, f3.z, f3.w };
        v16us ahu, alu;
#pragma unroll
        for (int j = 0; j < 16; ++j) {
            float x = av[j] * s;
            unsigned short hb = f32_to_bf16_rne(x);
            ahu[j] = hb;
            alu[j] = f32_to_bf16_rne(x - bf16_bits_to_f32(hb));
        }
        v16bf ah = __builtin_bit_cast(v16bf, ahu);
        v16bf al = __builtin_bit_cast(v16bf, alu);

        v16bf bh[4], bl[4];
#pragma unroll
        for (int u = 0; u < 4; ++u) {
            const unsigned short* bp = &ldsW[((t * 4 + u) * 32 + lane) * 16];
            bh[u] = __builtin_bit_cast(v16bf, *(const v16us*)bp);
            bl[u] = __builtin_bit_cast(v16bf, *(const v16us*)(bp + K * 64));
        }
        // Pass 1: Ah*Bh  (independent destinations back-to-back)
#pragma unroll
        for (int u = 0; u < 4; ++u)
            acc[u] = __builtin_amdgcn_wmma_f32_16x16x32_bf16(
                false, ah, false, bh[u], (short)0, acc[u], false, false);
        // Pass 2: Ah*Bl
#pragma unroll
        for (int u = 0; u < 4; ++u)
            acc[u] = __builtin_amdgcn_wmma_f32_16x16x32_bf16(
                false, ah, false, bl[u], (short)0, acc[u], false, false);
        // Pass 3: Al*Bh
#pragma unroll
        for (int u = 0; u < 4; ++u)
            acc[u] = __builtin_amdgcn_wmma_f32_16x16x32_bf16(
                false, al, false, bh[u], (short)0, acc[u], false, false);
    }

    // C/D layout: VGPR r -> M = r (lanes 0-15) / 8+r (lanes 16-31), N = lane&15
    const int ccol  = nb0 + (lane & 15);
    const int rbase = mrow0 + ((lane & 16) ? 8 : 0);
    if (mrow0 + 16 <= M) {
        // Fast path: full tile, unguarded stores (wave-uniform branch)
#pragma unroll
        for (int u = 0; u < 4; ++u) {
            int col  = ccol + u * 16;
            float bv = bias[col];
#pragma unroll
            for (int r = 0; r < 8; ++r)
                C[(size_t)(rbase + r) * N + col] = acc[u][r] + bv;
        }
    } else {
        // Ragged last tile
#pragma unroll
        for (int u = 0; u < 4; ++u) {
            int col  = ccol + u * 16;
            float bv = bias[col];
#pragma unroll
            for (int r = 0; r < 8; ++r) {
                int row = rbase + r;
                if (row < M) C[(size_t)row * N + col] = acc[u][r] + bv;
            }
        }
    }
}

// ---------------------------------------------------------------------------
// Utility kernels
// ---------------------------------------------------------------------------
__global__ void fill_kernel(float* __restrict__ p, int n, float v) {
    int i = blockIdx.x * blockDim.x + threadIdx.x;
    if (i < n) p[i] = v;
}

// n must be a multiple of 4
__global__ void fill4_kernel(float4* __restrict__ p, int n4, float v) {
    int i = blockIdx.x * blockDim.x + threadIdx.x;
    if (i < n4) { float4 q; q.x = v; q.y = v; q.z = v; q.w = v; p[i] = q; }
}

__global__ void deg_kernel(const int* __restrict__ src, const int* __restrict__ dst,
                           float* __restrict__ deg_u, float* __restrict__ deg_r, int e) {
    int i = blockIdx.x * blockDim.x + threadIdx.x;
    if (i < e) {
        atomicAdd(&deg_u[src[i]], 1.0f);
        atomicAdd(&deg_r[dst[i]], 1.0f);
    }
}

__global__ void rsqrt_clip_kernel(float* __restrict__ p, int n) {
    int i = blockIdx.x * blockDim.x + threadIdx.x;
    if (i < n) {
        float d = p[i];
        d = d < 1.0f ? 1.0f : d;
        p[i] = rsqrtf(d);
    }
}

// One wave per edge: agg[dst] += h[src] * rs_src[src]  (128-wide rows)
__global__ __launch_bounds__(256) void scatter_kernel(
    const float* __restrict__ h, const float* __restrict__ rs_src,
    const int* __restrict__ src, const int* __restrict__ dst,
    float* __restrict__ agg, int e)
{
    int widx = (int)((blockIdx.x * blockDim.x + threadIdx.x) >> 5);
    int lane = threadIdx.x & 31;
    if (widx >= e) return;
    int si = __builtin_amdgcn_readfirstlane(src[widx]);
    int di = __builtin_amdgcn_readfirstlane(dst[widx]);
    float s = rs_src[si];
    float4 v = *(const float4*)(h + (size_t)si * 128 + lane * 4);
    float* o = agg + (size_t)di * 128 + lane * 4;
    atomicAdd(o + 0, v.x * s);
    atomicAdd(o + 1, v.y * s);
    atomicAdd(o + 2, v.z * s);
    atomicAdd(o + 3, v.w * s);
}

// One wave per 64-wide row: in-place L2 normalize (eps = 1e-12)
__global__ __launch_bounds__(256) void l2norm_kernel(float* __restrict__ x, int rows) {
    int widx = (int)((blockIdx.x * blockDim.x + threadIdx.x) >> 5);
    int lane = threadIdx.x & 31;
    if (widx >= rows) return;
    float2 v = *(float2*)(x + (size_t)widx * 64 + lane * 2);
    float ss = v.x * v.x + v.y * v.y;
#pragma unroll
    for (int off = 16; off; off >>= 1) ss += __shfl_xor(ss, off, 32);
    float n = sqrtf(ss);
    n = n < 1e-12f ? 1e-12f : n;
    float inv = 1.0f / n;
    float2 o; o.x = v.x * inv; o.y = v.y * inv;
    *(float2*)(x + (size_t)widx * 64 + lane * 2) = o;
}

// One wave per edge: out[e] = dot(nu[iu[e]], nr[ir[e]]) over 64 dims
__global__ __launch_bounds__(256) void edge_dot_kernel(
    const float* __restrict__ nu, const float* __restrict__ nr,
    const int* __restrict__ iu, const int* __restrict__ ir,
    float* __restrict__ out, int e)
{
    int widx = (int)((blockIdx.x * blockDim.x + threadIdx.x) >> 5);
    int lane = threadIdx.x & 31;
    if (widx >= e) return;
    int u = __builtin_amdgcn_readfirstlane(iu[widx]);
    int r = __builtin_amdgcn_readfirstlane(ir[widx]);
    float2 a = *(const float2*)(nu + (size_t)u * 64 + lane * 2);
    float2 b = *(const float2*)(nr + (size_t)r * 64 + lane * 2);
    float d = a.x * b.x + a.y * b.y;
#pragma unroll
    for (int off = 16; off; off >>= 1) d += __shfl_xor(d, off, 32);
    if (lane == 0) out[widx] = d;
}

// ---------------------------------------------------------------------------
// Launch
// ---------------------------------------------------------------------------
extern "C" void kernel_launch(void* const* d_in, const int* in_sizes, int n_in,
                              void* d_out, int out_size, void* d_ws, size_t ws_size,
                              hipStream_t stream)
{
    const float* user_feat = (const float*)d_in[0];
    const float* repo_feat = (const float*)d_in[1];
    const float* W_u   = (const float*)d_in[2];
    const float* b_u   = (const float*)d_in[3];
    const float* W_rp  = (const float*)d_in[4];
    const float* b_rp  = (const float*)d_in[5];
    const float* W1_ur = (const float*)d_in[6];
    const float* b1_ur = (const float*)d_in[7];
    const float* W1_ru = (const float*)d_in[8];
    const float* b1_ru = (const float*)d_in[9];
    const float* W2_ur = (const float*)d_in[10];
    const float* b2_ur = (const float*)d_in[11];
    const float* W2_ru = (const float*)d_in[12];
    const float* b2_ru = (const float*)d_in[13];
    const int* e_src = (const int*)d_in[14];
    const int* e_dst = (const int*)d_in[15];
    const int* pos_u = (const int*)d_in[16];
    const int* pos_r = (const int*)d_in[17];
    const int* neg_u = (const int*)d_in[18];
    const int* neg_r = (const int*)d_in[19];
    float* out = (float*)d_out;

    // Workspace carve-up (peak ~206 MB)
    char* ws = (char*)d_ws;
    size_t off = 0;
    auto carve = [&](size_t bytes) -> float* {
        float* p = (float*)(ws + off);
        off += (bytes + 255) & ~(size_t)255;
        return p;
    };
    float* rs_u = carve((size_t)NU_N * 4);            // deg_u -> rsqrt
    float* rs_r = carve((size_t)NR_N * 4);            // deg_r -> rsqrt
    float* bufA = carve((size_t)NU_N * 128 * 4);      // h_user, later h2_user
    float* bufB = carve((size_t)NR_N * 128 * 4);      // h_repo, later h2_repo
    float* bufC = carve((size_t)NU_N * 128 * 4);      // agg scratch (reused 4x)
    float* bufD = carve((size_t)NR_N * 128 * 4);      // h1_repo
    float* bufE = carve((size_t)NU_N * 128 * 4);      // h1_user
    (void)ws_size; (void)in_sizes; (void)n_in; (void)out_size;

    const int GM_U = (NU_N + 127) / 128;   // 782
    const int GM_R = (NR_N + 127) / 128;   // 391
    const int EB   = (E_N + 7) / 8;        // scatter blocks (8 edges/block)
    const int ZR4  = NR_N * 128 / 4;       // float4 counts for agg zeroing
    const int ZU4  = NU_N * 128 / 4;

    // Degrees -> rsqrt(max(deg,1))  (shared by all 4 convs)
    fill_kernel<<<(NU_N + 255) / 256, 256, 0, stream>>>(rs_u, NU_N, 0.0f);
    fill_kernel<<<(NR_N + 255) / 256, 256, 0, stream>>>(rs_r, NR_N, 0.0f);
    deg_kernel<<<(E_N + 255) / 256, 256, 0, stream>>>(e_src, e_dst, rs_u, rs_r, E_N);
    rsqrt_clip_kernel<<<(NU_N + 255) / 256, 256, 0, stream>>>(rs_u, NU_N);
    rsqrt_clip_kernel<<<(NR_N + 255) / 256, 256, 0, stream>>>(rs_r, NR_N);

    // Input embeddings
    gemm_bias_wmma<128, 128><<<dim3(GM_U, 2), 256, 0, stream>>>(
        user_feat, nullptr, W_u, b_u, bufA, NU_N);
    gemm_bias_wmma<256, 128><<<dim3(GM_R, 2), 256, 0, stream>>>(
        repo_feat, nullptr, W_rp, b_rp, bufB, NR_N);

    // Layer 1: user -> repo
    fill4_kernel<<<(ZR4 + 255) / 256, 256, 0, stream>>>((float4*)bufC, ZR4, 0.0f);
    scatter_kernel<<<EB, 256, 0, stream>>>(bufA, rs_u, e_src, e_dst, bufC, E_N);
    gemm_bias_wmma<128, 128><<<dim3(GM_R, 2), 256, 0, stream>>>(
        bufC, rs_r, W1_ur, b1_ur, bufD, NR_N);
    // Layer 1: repo -> user
    fill4_kernel<<<(ZU4 + 255) / 256, 256, 0, stream>>>((float4*)bufC, ZU4, 0.0f);
    scatter_kernel<<<EB, 256, 0, stream>>>(bufB, rs_r, e_dst, e_src, bufC, E_N);
    gemm_bias_wmma<128, 128><<<dim3(GM_U, 2), 256, 0, stream>>>(
        bufC, rs_u, W1_ru, b1_ru, bufE, NU_N);

    // Layer 2: user -> repo (h2_repo into bufB, [NR,64])
    fill4_kernel<<<(ZR4 + 255) / 256, 256, 0, stream>>>((float4*)bufC, ZR4, 0.0f);
    scatter_kernel<<<EB, 256, 0, stream>>>(bufE, rs_u, e_src, e_dst, bufC, E_N);
    gemm_bias_wmma<128, 64><<<dim3(GM_R, 1), 256, 0, stream>>>(
        bufC, rs_r, W2_ur, b2_ur, bufB, NR_N);
    // Layer 2: repo -> user (h2_user into bufA, [NU,64])
    fill4_kernel<<<(ZU4 + 255) / 256, 256, 0, stream>>>((float4*)bufC, ZU4, 0.0f);
    scatter_kernel<<<EB, 256, 0, stream>>>(bufD, rs_r, e_dst, e_src, bufC, E_N);
    gemm_bias_wmma<128, 64><<<dim3(GM_U, 1), 256, 0, stream>>>(
        bufC, rs_u, W2_ru, b2_ru, bufA, NU_N);

    // Normalize + per-edge cosine scores
    l2norm_kernel<<<(NU_N + 7) / 8, 256, 0, stream>>>(bufA, NU_N);
    l2norm_kernel<<<(NR_N + 7) / 8, 256, 0, stream>>>(bufB, NR_N);
    edge_dot_kernel<<<(EP_N + 7) / 8, 256, 0, stream>>>(
        bufA, bufB, pos_u, pos_r, out, EP_N);
    edge_dot_kernel<<<(EN_N + 7) / 8, 256, 0, stream>>>(
        bufA, bufB, neg_u, neg_r, out + EP_N, EN_N);
}
